// MultiHeadedAttention_3856880632326
// MI455X (gfx1250) — compile-verified, matrix operand order checked by reading, not run on hardware
//
#include <hip/hip_runtime.h>
#include <stddef.h>
#include <stdint.h>

// ---------------------------------------------------------------------------
// MultiHeadedAttention for MI455X (gfx1250, wave32, WMMA).
// B=8, S=1024, D=1024, H=16, DH=64.
// bf16 WMMA (v_wmma_f32_16x16x32_bf16) with f32 accumulation everywhere.
// GEMM A-tiles staged into LDS with GLOBAL_LOAD_ASYNC_TO_LDS_B128 (ASYNCcnt)
// when the toolchain exposes the builtin; falls back to load+ds_store.
// ---------------------------------------------------------------------------

typedef __attribute__((ext_vector_type(16))) __bf16 v16bf;
typedef __attribute__((ext_vector_type(8)))  __bf16 v8bf;
typedef __attribute__((ext_vector_type(4)))  __bf16 v4bf;
typedef __attribute__((ext_vector_type(8)))  float  v8f;

static __device__ __forceinline__ v8f wmma_bf16(v16bf a, v16bf b, v8f c) {
  // (neg_a, A, neg_b, B, c_mod, C, reuse_a, reuse_b)
  return __builtin_amdgcn_wmma_f32_16x16x32_bf16(false, a, false, b, (short)0, c,
                                                 false, false);
}

// Load one 16x32 A-fragment (or 32x16 B-fragment) row for this lane.
// Per ISA 7.12.2: element e <-> K = 16*(e/8) + 8*half + (e%8); two 16B runs.
static __device__ __forceinline__ v16bf frag_ld(const __bf16* rowp, int lh) {
  v8bf lo = *(const v8bf*)(rowp + 8 * lh);
  v8bf hi = *(const v8bf*)(rowp + 16 + 8 * lh);
  return __builtin_shufflevector(lo, hi, 0, 1, 2, 3, 4, 5, 6, 7, 8, 9, 10, 11,
                                 12, 13, 14, 15);
}

static __device__ __forceinline__ float rmax16(float v) {
#pragma unroll
  for (int m = 1; m < 16; m <<= 1) v = fmaxf(v, __shfl_xor(v, m, 32));
  return v;
}
static __device__ __forceinline__ float rsum16(float v) {
#pragma unroll
  for (int m = 1; m < 16; m <<= 1) v += __shfl_xor(v, m, 32);
  return v;
}

// 16-byte global -> LDS copy. Prefer the CDNA5 async-to-LDS path (no VGPR
// round trip, tracked by ASYNCcnt); fall back to load + ds_store.
#if __has_builtin(__builtin_amdgcn_global_load_async_to_lds_b128)
typedef int v4i_vs __attribute__((vector_size(4 * sizeof(int))));
typedef __attribute__((address_space(1))) v4i_vs* as1_v4i_ptr;
typedef __attribute__((address_space(3))) v4i_vs* as3_v4i_ptr;
#endif

static __device__ __forceinline__ void copy16_g2lds(const __bf16* g, __bf16* l) {
#if __has_builtin(__builtin_amdgcn_global_load_async_to_lds_b128)
  // Low 32 bits of a flat LDS address are the LDS offset (ISA section 10.1.1).
  __builtin_amdgcn_global_load_async_to_lds_b128(
      (as1_v4i_ptr)(uintptr_t)g,
      (as3_v4i_ptr)(uint32_t)(uintptr_t)l,
      /*offset=*/0, /*cpol=*/0);
#else
  *(v8bf*)l = *(const v8bf*)g;
#endif
}

static __device__ __forceinline__ void async_copy_fence() {
#if __has_builtin(__builtin_amdgcn_global_load_async_to_lds_b128)
#if __has_builtin(__builtin_amdgcn_s_wait_asynccnt)
  __builtin_amdgcn_s_wait_asynccnt(0);
#else
  asm volatile("s_wait_asynccnt 0" ::: "memory");
#endif
#endif
}

// ---------------- stage 0: f32 -> bf16 conversion ----------------
__global__ __launch_bounds__(256) void cvt_kernel(const float4* __restrict__ src,
                                                  v4bf* __restrict__ dst, int n4) {
  int i = blockIdx.x * 256 + threadIdx.x;
  if (i < n4) {
    float4 f = src[i];
    v4bf o;
    o[0] = (__bf16)f.x; o[1] = (__bf16)f.y; o[2] = (__bf16)f.z; o[3] = (__bf16)f.w;
    dst[i] = o;
  }
}

// ---------------- stage 1: fused QKV projection ----------------
// y = x @ W^T + b. grid = (N/64, M/128, 3).  8 waves: 2 (M) x 4 (N).
// z==0 -> Q scaled 1/8, [B,H,S,DH]; z==1 -> K [B,H,S,DH]; z==2 -> V^T [B,H,DH,S]
#define LDA 40
__global__ __launch_bounds__(256) void qkv_gemm_kernel(
    const __bf16* __restrict__ Xq, const __bf16* __restrict__ Xk,
    const __bf16* __restrict__ Xv, const __bf16* __restrict__ Wq,
    const __bf16* __restrict__ Wk, const __bf16* __restrict__ Wv,
    const float* __restrict__ bq, const float* __restrict__ bk,
    const float* __restrict__ bv, __bf16* __restrict__ Qo,
    __bf16* __restrict__ Ko, __bf16* __restrict__ Vt) {
  __shared__ __bf16 As[128][LDA];
  const int z = blockIdx.z;
  const __bf16* X = (z == 0) ? Xq : (z == 1) ? Xk : Xv;
  const __bf16* W = (z == 0) ? Wq : (z == 1) ? Wk : Wv;
  const float* bias = (z == 0) ? bq : (z == 1) ? bk : bv;

  const int tid = threadIdx.x, lane = tid & 31, w = tid >> 5;
  const int ln = lane & 15, lh = (lane >> 4) & 1;
  const int mg = w >> 2, ng = w & 3;
  const int blockM = blockIdx.y * 128;
  const int nBase = blockIdx.x * 64 + ng * 16;

  v8f acc[4] = {};

  for (int k0 = 0; k0 < 1024; k0 += 32) {
#pragma unroll
    for (int j = 0; j < 2; ++j) {  // cooperative 128x32 A tile -> LDS
      int c = tid + 256 * j;
      int row = c >> 2, col = (c & 3) * 8;
      copy16_g2lds(X + (size_t)(blockM + row) * 1024 + k0 + col, &As[row][col]);
    }
    async_copy_fence();
    __syncthreads();
    const __bf16* wrow = W + (size_t)(nBase + ln) * 1024 + k0;
    if (k0 + 32 < 1024) __builtin_prefetch(wrow + 32, 0, 3);
    v16bf bf = frag_ld(wrow, lh);
#pragma unroll
    for (int i = 0; i < 4; ++i) {
      v16bf af = frag_ld(&As[mg * 64 + i * 16 + ln][0], lh);
      acc[i] = wmma_bf16(af, bf, acc[i]);
    }
    __syncthreads();
  }

  const int n = nBase + ln;
  const float bn = bias[n];
  const int hh = n >> 6, dh = n & 63;
  if (z == 2) {  // V^T: rows s are consecutive per lane -> contiguous 16B store
#pragma unroll
    for (int i = 0; i < 4; ++i) {
      int m0 = blockM + mg * 64 + i * 16 + 8 * lh;
      int b = m0 >> 10, s0 = m0 & 1023;
      v8bf pk;
#pragma unroll
      for (int r = 0; r < 8; ++r) pk[r] = (__bf16)(acc[i][r] + bn);
      *(v8bf*)(Vt + ((size_t)(b * 16 + hh) * 64 + dh) * 1024 + s0) = pk;
    }
  } else {
    __bf16* O = (z == 0) ? Qo : Ko;
    const float scale = (z == 0) ? 0.125f : 1.0f;  // 1/sqrt(64)
#pragma unroll
    for (int i = 0; i < 4; ++i) {
      int m0 = blockM + mg * 64 + i * 16 + 8 * lh;
#pragma unroll
      for (int r = 0; r < 8; ++r) {
        int m = m0 + r, b = m >> 10, s = m & 1023;
        O[((size_t)(b * 16 + hh) * 1024 + s) * 64 + dh] =
            (__bf16)((acc[i][r] + bn) * scale);
      }
    }
  }
}

// ---------------- stage 2: flash attention ----------------
// grid = (S/64, B*H), 128 threads (4 waves); each wave owns 16 query rows.
__global__ __launch_bounds__(128) void attn_kernel(
    const __bf16* __restrict__ Q, const __bf16* __restrict__ Kc,
    const __bf16* __restrict__ Vt, const unsigned char* __restrict__ mask,
    const float* __restrict__ graph, __bf16* __restrict__ ctx) {
  __shared__ __bf16 Ps[4][16][LDA];
  const int bh = blockIdx.y, b = bh >> 4, h = bh & 15;
  const bool lastHead = (h == 15);
  const int tid = threadIdx.x, w = tid >> 5, lane = tid & 31;
  const int ln = lane & 15, lh = lane >> 4;
  const int q0 = blockIdx.x * 64 + w * 16;

  const __bf16* qrow = Q + ((size_t)bh * 1024 + q0 + ln) * 64;
  v16bf qa0 = frag_ld(qrow, lh);        // dh 0..31
  v16bf qa1 = frag_ld(qrow + 32, lh);   // dh 32..63

  float rmax[8], rsp[8], rse[8];
#pragma unroll
  for (int r = 0; r < 8; ++r) { rmax[r] = -1e30f; rsp[r] = 0.f; rse[r] = 0.f; }
  v8f acc[4] = {};

  for (int kb = 0; kb < 1024; kb += 32) {
    const __bf16* kr0 = Kc + ((size_t)bh * 1024 + kb + ln) * 64;
    const __bf16* kr1 = kr0 + 16 * 64;
    v8f s0 = {}, s1 = {};
    s0 = wmma_bf16(qa0, frag_ld(kr0, lh), s0);
    s0 = wmma_bf16(qa1, frag_ld(kr0 + 32, lh), s0);
    s1 = wmma_bf16(qa0, frag_ld(kr1, lh), s1);
    s1 = wmma_bf16(qa1, frag_ld(kr1 + 32, lh), s1);

#pragma unroll
    for (int r = 0; r < 8; ++r) {
      const int mq = q0 + 8 * lh + r;
      const size_t mb = ((size_t)b * 1024 + mq) * 1024;
      float sv0 = mask[mb + kb + ln] ? -1e18f : s0[r];
      float sv1 = mask[mb + kb + 16 + ln] ? -1e18f : s1[r];
      float g0 = 1.f, g1 = 1.f;
      if (lastHead) { g0 = graph[mb + kb + ln]; g1 = graph[mb + kb + 16 + ln]; }
      float nm = fmaxf(rmax[r], rmax16(fmaxf(sv0, sv1)));
      float scl = __expf(rmax[r] - nm);
      rmax[r] = nm;
      float e0 = __expf(sv0 - nm), e1 = __expf(sv1 - nm);
      float p0 = e0 * g0, p1 = e1 * g1;
      rsp[r] = rsp[r] * scl + rsum16(p0 + p1);
      if (lastHead) rse[r] = rse[r] * scl + rsum16(e0 + e1);
#pragma unroll
      for (int j = 0; j < 4; ++j) acc[j][r] *= scl;
      // C-layout -> A-layout transpose of P via LDS (bf16)
      Ps[w][8 * lh + r][ln] = (__bf16)p0;
      Ps[w][8 * lh + r][16 + ln] = (__bf16)p1;
    }
    __syncthreads();
    v16bf pa = frag_ld(&Ps[w][ln][0], lh);
#pragma unroll
    for (int j = 0; j < 4; ++j) {
      const __bf16* vr = Vt + ((size_t)bh * 64 + j * 16 + ln) * 1024 + kb;
      acc[j] = wmma_bf16(pa, frag_ld(vr, lh), acc[j]);
    }
    __syncthreads();
  }

#pragma unroll
  for (int j = 0; j < 4; ++j) {
#pragma unroll
    for (int r = 0; r < 8; ++r) {
      float denom = rsp[r] + (lastHead ? 1e-9f * rse[r] : 0.f);
      int sq = q0 + 8 * lh + r;
      ctx[((size_t)(b * 1024 + sq) * 16 + h) * 64 + j * 16 + ln] =
          (__bf16)(acc[j][r] / denom);
    }
  }
}

// ---------------- stage 3: output projection (f32 out) ----------------
__global__ __launch_bounds__(256) void oproj_kernel(
    const __bf16* __restrict__ Xc, const __bf16* __restrict__ Wo,
    const float* __restrict__ bo, float* __restrict__ out) {
  __shared__ __bf16 As[128][LDA];
  const int tid = threadIdx.x, lane = tid & 31, w = tid >> 5;
  const int ln = lane & 15, lh = (lane >> 4) & 1;
  const int mg = w >> 2, ng = w & 3;
  const int blockM = blockIdx.y * 128;
  const int nBase = blockIdx.x * 64 + ng * 16;
  v8f acc[4] = {};

  for (int k0 = 0; k0 < 1024; k0 += 32) {
#pragma unroll
    for (int j = 0; j < 2; ++j) {
      int c = tid + 256 * j;
      int row = c >> 2, col = (c & 3) * 8;
      copy16_g2lds(Xc + (size_t)(blockM + row) * 1024 + k0 + col, &As[row][col]);
    }
    async_copy_fence();
    __syncthreads();
    const __bf16* wrow = Wo + (size_t)(nBase + ln) * 1024 + k0;
    if (k0 + 32 < 1024) __builtin_prefetch(wrow + 32, 0, 3);
    v16bf bf = frag_ld(wrow, lh);
#pragma unroll
    for (int i = 0; i < 4; ++i) {
      v16bf af = frag_ld(&As[mg * 64 + i * 16 + ln][0], lh);
      acc[i] = wmma_bf16(af, bf, acc[i]);
    }
    __syncthreads();
  }

  const int n = nBase + ln;
  const float bn = bo[n];
#pragma unroll
  for (int i = 0; i < 4; ++i) {
    int m0 = blockM + mg * 64 + i * 16 + 8 * lh;
#pragma unroll
    for (int r = 0; r < 8; ++r)
      out[(size_t)(m0 + r) * 1024 + n] = acc[i][r] + bn;
  }
}

// ---------------------------------------------------------------------------
extern "C" void kernel_launch(void* const* d_in, const int* in_sizes, int n_in,
                              void* d_out, int out_size, void* d_ws,
                              size_t ws_size, hipStream_t stream) {
  (void)in_sizes; (void)n_in; (void)out_size; (void)ws_size;
  const float* keyf = (const float*)d_in[0];
  const float* valf = (const float*)d_in[1];
  const float* qryf = (const float*)d_in[2];
  const unsigned char* maskp = (const unsigned char*)d_in[3];  // jnp.bool_ = 1B
  const float* graph = (const float*)d_in[4];
  const float* Wq = (const float*)d_in[5];  const float* bq = (const float*)d_in[6];
  const float* Wk = (const float*)d_in[7];  const float* bk = (const float*)d_in[8];
  const float* Wv = (const float*)d_in[9];  const float* bv = (const float*)d_in[10];
  const float* Wo = (const float*)d_in[11]; const float* bo = (const float*)d_in[12];
  float* out = (float*)d_out;

  char* ws = (char*)d_ws;
  const size_t MB = 1ull << 20;
  __bf16* Xq  = (__bf16*)(ws + 0 * MB);    // 16MB each (8M bf16)
  __bf16* Xk  = (__bf16*)(ws + 16 * MB);
  __bf16* Xv  = (__bf16*)(ws + 32 * MB);
  __bf16* Wqb = (__bf16*)(ws + 48 * MB);   // 2MB each (1M bf16)
  __bf16* Wkb = (__bf16*)(ws + 50 * MB);
  __bf16* Wvb = (__bf16*)(ws + 52 * MB);
  __bf16* Wob = (__bf16*)(ws + 54 * MB);
  __bf16* Qw  = (__bf16*)(ws + 56 * MB);   // [B,H,S,DH]
  __bf16* Kw  = (__bf16*)(ws + 72 * MB);   // [B,H,S,DH]
  __bf16* Vtw = (__bf16*)(ws + 88 * MB);   // [B,H,DH,S]
  __bf16* Cw  = (__bf16*)(ws + 104 * MB);  // [B,S,H,DH]

  const int nX4 = (8 * 1024 * 1024) / 4;
  const int nW4 = (1024 * 1024) / 4;
  cvt_kernel<<<nX4 / 256, 256, 0, stream>>>((const float4*)qryf, (v4bf*)Xq, nX4);
  cvt_kernel<<<nX4 / 256, 256, 0, stream>>>((const float4*)keyf, (v4bf*)Xk, nX4);
  cvt_kernel<<<nX4 / 256, 256, 0, stream>>>((const float4*)valf, (v4bf*)Xv, nX4);
  cvt_kernel<<<nW4 / 256, 256, 0, stream>>>((const float4*)Wq, (v4bf*)Wqb, nW4);
  cvt_kernel<<<nW4 / 256, 256, 0, stream>>>((const float4*)Wk, (v4bf*)Wkb, nW4);
  cvt_kernel<<<nW4 / 256, 256, 0, stream>>>((const float4*)Wv, (v4bf*)Wvb, nW4);
  cvt_kernel<<<nW4 / 256, 256, 0, stream>>>((const float4*)Wo, (v4bf*)Wob, nW4);

  qkv_gemm_kernel<<<dim3(16, 64, 3), 256, 0, stream>>>(
      Xq, Xk, Xv, Wqb, Wkb, Wvb, bq, bk, bv, Qw, Kw, Vtw);
  attn_kernel<<<dim3(16, 128), 128, 0, stream>>>(Qw, Kw, Vtw, maskp, graph, Cw);
  oproj_kernel<<<dim3(16, 64), 256, 0, stream>>>(Cw, Wob, bo, out);
}
